// SentimentClassifier_11510512353360
// MI455X (gfx1250) — compile-verified, matrix operand order checked by reading, not run on hardware
//
#include <hip/hip_runtime.h>

// ---------------- problem constants ----------------
#define BB   16
#define SS   512
#define DD   768
#define HH   12
#define HD   64
#define FF_  3072
#define LL   6
#define NC_  3
#define MM   (BB * SS)          // 8192 tokens

// ---------------- vector types ----------------
typedef __attribute__((ext_vector_type(16))) _Float16      v16h;
typedef __attribute__((ext_vector_type(8)))  float         v8f;
typedef __attribute__((ext_vector_type(4)))  unsigned int  u32x4;
typedef __attribute__((ext_vector_type(4)))  int           i32x4;

union Frag16 { u32x4 q[2]; v16h h; };

// ---------------- async global -> LDS staging (CDNA5) ----------------
// GLOBAL_LOAD_ASYNC_TO_LDS_B128: per lane, LDS[ldst] = MEM[gsrc] (16 bytes),
// tracked with ASYNCcnt. Builtin takes int4* pointers (global src, LDS dst).
// Falls back to a b128 register copy if the builtins are not declared.
#if defined(__AMDGCN__) &&                                        \
    __has_builtin(__builtin_amdgcn_global_load_async_to_lds_b128) && \
    __has_builtin(__builtin_amdgcn_s_wait_asynccnt)
#define USE_ASYNC_LDS 1
typedef __attribute__((address_space(1))) i32x4 as1_i32x4;
typedef __attribute__((address_space(3))) i32x4 as3_i32x4;
#else
#define USE_ASYNC_LDS 0
#endif

__device__ inline void cp_b128_to_lds(const _Float16* gsrc, _Float16* ldst) {
#if USE_ASYNC_LDS
  __builtin_amdgcn_global_load_async_to_lds_b128(
      (as1_i32x4*)gsrc, (as3_i32x4*)ldst, 0, 0);
#else
  *(u32x4*)ldst = *(const u32x4*)gsrc;
#endif
}

__device__ inline void cp_lds_wait() {
#if USE_ASYNC_LDS
  __builtin_amdgcn_s_wait_asynccnt(0);
#endif
}

// Load one 16x32 f16 fragment (A-layout per CDNA5 ISA 7.12.2):
// lanes 0-15: row=lane,    halves K=[0..7] then K=[16..23]
// lanes 16-31: row=lane-16, halves K=[8..15] then K=[24..31]
__device__ inline v16h load_frag(const _Float16* base, int strideHalves, int lane) {
  int r  = lane & 15;
  int hi = (lane >> 4) & 1;
  const _Float16* p = base + (size_t)r * strideHalves + hi * 8;
  Frag16 f;
  f.q[0] = *(const u32x4*)(p);
  f.q[1] = *(const u32x4*)(p + 16);
  return f.h;
}

__device__ inline v8f wmma_f16(v16h a, v16h b, v8f c) {
  return __builtin_amdgcn_wmma_f32_16x16x32_f16(false, a, false, b, (short)0, c,
                                                false, false);
}

// ---------------- f32 -> f16 convert + transpose (weights) ----------------
// src: [layers][Kd][Nd] f32 row-major;  dst: [layers][Nd][Kd] f16 (N-major).
__global__ void convert_transpose_kernel(const float* __restrict__ src,
                                         _Float16* __restrict__ dst,
                                         int Kd, int Nd, int layers) {
  size_t total = (size_t)layers * Kd * Nd;
  size_t stride = (size_t)gridDim.x * blockDim.x;
  for (size_t i = (size_t)blockIdx.x * blockDim.x + threadIdx.x; i < total;
       i += stride) {
    size_t per = (size_t)Kd * Nd;
    int l = (int)(i / per);
    size_t r = i % per;
    int kk = (int)(r / Nd);
    int n  = (int)(r % Nd);
    dst[((size_t)l * Nd + n) * Kd + kk] = (_Float16)src[i];
  }
}

// ---------------- V head-transpose: [B,S,H*HD] -> [B,H,HD,S] ----------------
__global__ void v_transpose_kernel(const _Float16* __restrict__ v,
                                   _Float16* __restrict__ vt) {
  size_t i = (size_t)blockIdx.x * blockDim.x + threadIdx.x;
  if (i < (size_t)MM * DD) {
    int d = (int)(i % DD);
    size_t t = i / DD;
    int s = (int)(t % SS);
    int b = (int)(t / SS);
    int h = d / HD, dd = d % HD;
    vt[(((size_t)b * HH + h) * HD + dd) * SS + s] = v[i];
  }
}

// ---------------- ragged segment merge + pos emb ----------------
__global__ __launch_bounds__(256) void merge_kernel(
    const int* __restrict__ ids, const int* __restrict__ segs,
    const float* __restrict__ emb, const float* __restrict__ pos,
    float* __restrict__ out) {
  int sOut = blockIdx.x, b = blockIdx.y;
  int d0 = threadIdx.x * 3;                    // 768 = 256 * 3
  float a0 = 0.f, a1 = 0.f, a2 = 0.f;
  int cnt = 0;
  const int* srow = segs + (size_t)b * SS;
  const int* irow = ids  + (size_t)b * SS;
  for (int s = 0; s < SS; ++s) {
    if (srow[s] == sOut) {                     // uniform branch across block
      const float* e = emb + (size_t)irow[s] * DD + d0;
      a0 += e[0]; a1 += e[1]; a2 += e[2]; ++cnt;
    }
  }
  if (cnt == 0) {
    const float* e = emb + d0;                 // emb_table[0]
    a0 = e[0]; a1 = e[1]; a2 = e[2];
  }
  const float* pp = pos + (size_t)sOut * DD + d0;
  float* o = out + ((size_t)b * SS + sOut) * DD + d0;
  o[0] = a0 + pp[0]; o[1] = a1 + pp[1]; o[2] = a2 + pp[2];
}

// ---------------- (a + b) -> LayerNorm -> f32 + f16 ----------------
__global__ __launch_bounds__(256) void add_ln_kernel(
    const float* __restrict__ a, const float* __restrict__ bres,
    const float* __restrict__ g, const float* __restrict__ be,
    float* __restrict__ outF, _Float16* __restrict__ outH) {
  __shared__ float rs[256], rq[256];
  int t = blockIdx.x, tid = threadIdx.x;
  int d0 = tid * 3;
  const float* pa = a + (size_t)t * DD + d0;
  float v0 = pa[0], v1 = pa[1], v2 = pa[2];
  if (bres) {
    const float* pb = bres + (size_t)t * DD + d0;
    v0 += pb[0]; v1 += pb[1]; v2 += pb[2];
  }
  rs[tid] = v0 + v1 + v2;
  rq[tid] = v0 * v0 + v1 * v1 + v2 * v2;
  __syncthreads();
  for (int off = 128; off > 0; off >>= 1) {
    if (tid < off) { rs[tid] += rs[tid + off]; rq[tid] += rq[tid + off]; }
    __syncthreads();
  }
  float mean = rs[0] * (1.0f / DD);
  float var  = rq[0] * (1.0f / DD) - mean * mean;
  float inv  = rsqrtf(var + 1e-12f);
  float o0 = (v0 - mean) * inv * g[d0 + 0] + be[d0 + 0];
  float o1 = (v1 - mean) * inv * g[d0 + 1] + be[d0 + 1];
  float o2 = (v2 - mean) * inv * g[d0 + 2] + be[d0 + 2];
  size_t base = (size_t)t * DD + d0;
  if (outF) { outF[base + 0] = o0; outF[base + 1] = o1; outF[base + 2] = o2; }
  if (outH) {
    outH[base + 0] = (_Float16)o0; outH[base + 1] = (_Float16)o1;
    outH[base + 2] = (_Float16)o2;
  }
}

// ---------------- WMMA GEMM:  C[M,N] = act(A[M,K] @ B^T[N,K] + bias) ---------
// 256 threads = 8 waves (4 x 2), tile 128x64, K-step 64.
// Each wave owns a 32x32 register tile (4 f32 accumulators, 8 WMMA / K-step).
#define BM 128
#define BN 64
#define BK 64
__global__ __launch_bounds__(256) void gemm_f16_kernel(
    const _Float16* __restrict__ A,      // [M][K]
    const _Float16* __restrict__ Bt,     // [N][K] pre-transposed
    const float* __restrict__ bias,
    float* __restrict__ outF32, _Float16* __restrict__ outF16,
    int M, int N, int K, int act) {
  __shared__ _Float16 As[BM][BK];        // 16 KB
  __shared__ _Float16 Bs[BN][BK];        //  8 KB
  int tid = threadIdx.x;
  int wave = tid >> 5, lane = tid & 31;
  int m0 = blockIdx.x * BM, n0 = blockIdx.y * BN;
  int mw = (wave & 3) * 32;              // wave row offset (4 waves along M)
  int nw = (wave >> 2) * 32;             // wave col offset (2 waves along N)
  v8f c00 = {}, c01 = {}, c10 = {}, c11 = {};
  for (int k0 = 0; k0 < K; k0 += BK) {
#pragma unroll
    for (int j = 0; j < 4; ++j) {        // A: 1024 b128 / 256 thr = 4 each
      int vv = tid + 256 * j;
      int r = vv >> 3, cc = (vv & 7) * 8;
      cp_b128_to_lds(&A[(size_t)(m0 + r) * K + k0 + cc], &As[r][cc]);
    }
#pragma unroll
    for (int j = 0; j < 2; ++j) {        // B: 512 b128 / 256 thr = 2 each
      int vv = tid + 256 * j;
      int r = vv >> 3, cc = (vv & 7) * 8;
      cp_b128_to_lds(&Bt[(size_t)(n0 + r) * K + k0 + cc], &Bs[r][cc]);
    }
    if (k0 + BK < K)
      __builtin_prefetch(&A[(size_t)(m0 + (tid & 127)) * K + k0 + BK], 0, 3);
    cp_lds_wait();
    __syncthreads();
#pragma unroll
    for (int kc = 0; kc < 2; ++kc) {     // two 32-wide chunks per staged tile
      v16h a0 = load_frag(&As[mw][kc * 32],      BK, lane);
      v16h a1 = load_frag(&As[mw + 16][kc * 32], BK, lane);
      v16h b0 = load_frag(&Bs[nw][kc * 32],      BK, lane);
      v16h b1 = load_frag(&Bs[nw + 16][kc * 32], BK, lane);
      c00 = wmma_f16(a0, b0, c00);
      c01 = wmma_f16(a0, b1, c01);
      c10 = wmma_f16(a1, b0, c10);
      c11 = wmma_f16(a1, b1, c11);
    }
    __syncthreads();
  }
  // epilogue: lane = column within subtile, VGPR r = row (+8 upper half-wave)
  int ncl = lane & 15;
  int hi  = (lane >> 4) ? 8 : 0;
  auto store_tile = [&](v8f c, int subm, int subn) {
    int mb = m0 + mw + subm + hi;
    int nc = n0 + nw + subn + ncl;
    float bv = bias ? bias[nc] : 0.0f;
#pragma unroll
    for (int r = 0; r < 8; ++r) {
      float val = c[r] + bv;
      if (act == 1) val = 0.5f * val * (1.0f + erff(val * 0.70710678118f));
      size_t off = (size_t)(mb + r) * N + nc;
      if (outF32) outF32[off] = val;
      if (outF16) outF16[off] = (_Float16)val;
    }
  };
  store_tile(c00, 0, 0);
  store_tile(c01, 0, 16);
  store_tile(c10, 16, 0);
  store_tile(c11, 16, 16);
}

// ---------------- attention: one wave per (b, h, 16-query tile) ----------------
// vt is [B,H,HD,S] so the ctx B-tile staging is a straight vector copy.
__global__ __launch_bounds__(32) void attn_kernel(
    const _Float16* __restrict__ q, const _Float16* __restrict__ k,
    const _Float16* __restrict__ vt, const int* __restrict__ mask,
    _Float16* __restrict__ ctx) {
  __shared__ _Float16 Qs[16][HD];       //  2 KB
  __shared__ _Float16 Ks[16][HD];       //  2 KB (N-major B tile == key rows)
  __shared__ _Float16 Vt[16][32];       //  1 KB (rows = dims, from vt layout)
  __shared__ float    Sc[16][SS];       // 32 KB scores
  __shared__ _Float16 Ph[16][SS];       // 16 KB probabilities f16
  int lane = threadIdx.x;
  int qt = blockIdx.x, h = blockIdx.y, b = blockIdx.z;
  size_t qbase = ((size_t)b * SS + qt * 16) * DD + h * HD;

  for (int i = lane; i < 16 * 8; i += 32) {       // 16 rows x 8 b128
    int r = i >> 3, gofs = (i & 7) * 8;
    cp_b128_to_lds(&q[qbase + (size_t)r * DD + gofs], &Qs[r][gofs]);
  }
  cp_lds_wait();
  __syncthreads();

  // ---- scores: S[16, 512] = Q (16x64) @ K^T, scaled + mask bias ----
  for (int kt = 0; kt < SS / 16; ++kt) {
    size_t kbase = ((size_t)b * SS + kt * 16) * DD + h * HD;
    for (int i = lane; i < 16 * 8; i += 32) {
      int r = i >> 3, gofs = (i & 7) * 8;
      cp_b128_to_lds(&k[kbase + (size_t)r * DD + gofs], &Ks[r][gofs]);
    }
    cp_lds_wait();
    __syncthreads();
    v8f c = {};
#pragma unroll
    for (int ks = 0; ks < 2; ++ks) {
      v16h fa = load_frag(&Qs[0][ks * 32], HD, lane);
      v16h fb = load_frag(&Ks[0][ks * 32], HD, lane);
      c = wmma_f16(fa, fb, c);
    }
    int ncol = lane & 15;
    int mrow = (lane >> 4) ? 8 : 0;
    float mb = (mask[b * SS + kt * 16 + ncol] == 0) ? -1e9f : 0.0f;
#pragma unroll
    for (int r = 0; r < 8; ++r)
      Sc[mrow + r][kt * 16 + ncol] = c[r] * 0.125f + mb;   // 1/sqrt(64)
    __syncthreads();
  }

  // ---- softmax over 512 per row (2 lanes per row, combine via shfl) ----
  {
    int row = lane & 15, c0 = (lane >> 4) * 256;
    float mx = -1e30f;
    for (int j = 0; j < 256; ++j) mx = fmaxf(mx, Sc[row][c0 + j]);
    mx = fmaxf(mx, __shfl_xor(mx, 16, 32));
    float sum = 0.f;
    for (int j = 0; j < 256; ++j) {
      float e = __expf(Sc[row][c0 + j] - mx);
      Sc[row][c0 + j] = e; sum += e;
    }
    sum += __shfl_xor(sum, 16, 32);
    float inv = 1.0f / sum;
    for (int j = 0; j < 256; ++j)
      Ph[row][c0 + j] = (_Float16)(Sc[row][c0 + j] * inv);
  }
  __syncthreads();

  // ---- ctx: (16x64) = P (16x512) @ V (512x64), 4 column subtiles ----
  size_t obase = ((size_t)b * SS + qt * 16) * DD + h * HD;
  for (int nt = 0; nt < 4; ++nt) {
    size_t vtbase = (((size_t)b * HH + h) * HD + nt * 16) * SS;
    v8f c = {};
    for (int ks = 0; ks < SS / 32; ++ks) {
#pragma unroll
      for (int j = 0; j < 2; ++j) {     // Vt: 64 b128 / 32 lanes = 2 each
        int vv = lane + 32 * j;
        int r = vv >> 2, cc = (vv & 3) * 8;
        cp_b128_to_lds(&vt[vtbase + (size_t)r * SS + ks * 32 + cc], &Vt[r][cc]);
      }
      cp_lds_wait();
      __syncthreads();
      v16h fa = load_frag(&Ph[0][ks * 32], SS, lane);
      v16h fb = load_frag(&Vt[0][0], 32, lane);
      c = wmma_f16(fa, fb, c);
      __syncthreads();
    }
    int ncol = lane & 15, mrow = (lane >> 4) ? 8 : 0;
#pragma unroll
    for (int r = 0; r < 8; ++r)
      ctx[obase + (size_t)(mrow + r) * DD + nt * 16 + ncol] = (_Float16)c[r];
  }
}

// ---------------- classifier head ----------------
__global__ __launch_bounds__(256) void cls_pre_kernel(
    const float* __restrict__ x, const float* __restrict__ W,
    const float* __restrict__ bias, float* __restrict__ out) {
  int b = blockIdx.x;
  const float* row = x + (size_t)b * SS * DD;     // token 0 of batch b
  for (int cI = threadIdx.x; cI < DD; cI += 256) {
    float acc = bias[cI];
    for (int kk = 0; kk < DD; ++kk) acc += row[kk] * W[(size_t)kk * DD + cI];
    out[(size_t)b * DD + cI] = fmaxf(acc, 0.0f);
  }
}

__global__ void cls_out_kernel(const float* __restrict__ pre,
                               const float* __restrict__ W,
                               const float* __restrict__ bias,
                               float* __restrict__ out) {
  int i = threadIdx.x;
  if (i < BB * NC_) {
    int b = i / NC_, cI = i % NC_;
    float acc = bias[cI];
    for (int kk = 0; kk < DD; ++kk)
      acc += pre[(size_t)b * DD + kk] * W[kk * NC_ + cI];
    out[i] = acc;
  }
}

// ---------------- host launcher ----------------
extern "C" void kernel_launch(void* const* d_in, const int* in_sizes, int n_in,
                              void* d_out, int out_size, void* d_ws, size_t ws_size,
                              hipStream_t stream) {
  const int*   input_ids = (const int*)d_in[0];
  const int*   segment_ids = (const int*)d_in[1];
  const int*   attention_mask = (const int*)d_in[2];
  // d_in[3] = label_ids (unused)
  const float* emb_table = (const float*)d_in[4];
  const float* pos_emb   = (const float*)d_in[5];
  const float* emb_ln_s  = (const float*)d_in[6];
  const float* emb_ln_b  = (const float*)d_in[7];
  const float* Wq = (const float*)d_in[8];   const float* bq = (const float*)d_in[9];
  const float* Wk = (const float*)d_in[10];  const float* bk = (const float*)d_in[11];
  const float* Wv = (const float*)d_in[12];  const float* bv = (const float*)d_in[13];
  const float* Wo = (const float*)d_in[14];  const float* bo = (const float*)d_in[15];
  const float* ln1_s = (const float*)d_in[16]; const float* ln1_b = (const float*)d_in[17];
  const float* W1 = (const float*)d_in[18];  const float* b1 = (const float*)d_in[19];
  const float* W2 = (const float*)d_in[20];  const float* b2 = (const float*)d_in[21];
  const float* ln2_s = (const float*)d_in[22]; const float* ln2_b = (const float*)d_in[23];
  const float* pre_W = (const float*)d_in[24]; const float* pre_b = (const float*)d_in[25];
  const float* cls_W = (const float*)d_in[26]; const float* cls_b = (const float*)d_in[27];
  float* logits = (float*)d_out;

  // ---- workspace carve ----
  char* w = (char*)d_ws;
  auto carve = [&](size_t bytes) {
    void* p = (void*)w;
    w += (bytes + 255) & ~(size_t)255;
    return p;
  };
  float*     tmp  = (float*)    carve((size_t)MM * DD * 4);   // merged / GEMM f32 out
  float*     x    = (float*)    carve((size_t)MM * DD * 4);   // residual stream f32
  _Float16*  xh   = (_Float16*) carve((size_t)MM * DD * 2);
  _Float16*  qh   = (_Float16*) carve((size_t)MM * DD * 2);
  _Float16*  kh   = (_Float16*) carve((size_t)MM * DD * 2);
  _Float16*  vh   = (_Float16*) carve((size_t)MM * DD * 2);
  _Float16*  vth  = (_Float16*) carve((size_t)MM * DD * 2);   // [B,H,HD,S]
  _Float16*  ctxh = (_Float16*) carve((size_t)MM * DD * 2);
  _Float16*  hh   = (_Float16*) carve((size_t)MM * FF_ * 2);
  size_t nWdd = (size_t)LL * DD * DD;
  size_t nWff = (size_t)LL * DD * FF_;
  _Float16* WqH = (_Float16*) carve(nWdd * 2);   // all stored [N][K]
  _Float16* WkH = (_Float16*) carve(nWdd * 2);
  _Float16* WvH = (_Float16*) carve(nWdd * 2);
  _Float16* WoH = (_Float16*) carve(nWdd * 2);
  _Float16* W1H = (_Float16*) carve(nWff * 2);
  _Float16* W2H = (_Float16*) carve(nWff * 2);
  float* pre_out = (float*) carve((size_t)BB * DD * 4);

  // ---- weights -> f16, transposed to [N][K] ----
  convert_transpose_kernel<<<4096, 256, 0, stream>>>(Wq, WqH, DD, DD, LL);
  convert_transpose_kernel<<<4096, 256, 0, stream>>>(Wk, WkH, DD, DD, LL);
  convert_transpose_kernel<<<4096, 256, 0, stream>>>(Wv, WvH, DD, DD, LL);
  convert_transpose_kernel<<<4096, 256, 0, stream>>>(Wo, WoH, DD, DD, LL);
  convert_transpose_kernel<<<4096, 256, 0, stream>>>(W1, W1H, DD, FF_, LL);
  convert_transpose_kernel<<<4096, 256, 0, stream>>>(W2, W2H, FF_, DD, LL);

  // ---- embeddings: segment merge + pos, then LN ----
  merge_kernel<<<dim3(SS, BB), 256, 0, stream>>>(input_ids, segment_ids,
                                                 emb_table, pos_emb, tmp);
  add_ln_kernel<<<MM, 256, 0, stream>>>(tmp, nullptr, emb_ln_s, emb_ln_b, x, xh);

  // ---- encoder layers ----
  dim3 gD(MM / BM, DD / BN);    // 64 x 12
  dim3 gF(MM / BM, FF_ / BN);   // 64 x 48
  dim3 gA(SS / 16, HH, BB);     // 32 x 12 x 16
  int vtBlocks = (int)(((size_t)MM * DD + 255) / 256);
  for (int l = 0; l < LL; ++l) {
    const _Float16* wq = WqH + (size_t)l * DD * DD;
    const _Float16* wk = WkH + (size_t)l * DD * DD;
    const _Float16* wv = WvH + (size_t)l * DD * DD;
    const _Float16* wo = WoH + (size_t)l * DD * DD;
    const _Float16* w1 = W1H + (size_t)l * DD * FF_;
    const _Float16* w2 = W2H + (size_t)l * FF_ * DD;
    gemm_f16_kernel<<<gD, 256, 0, stream>>>(xh, wq, bq + (size_t)l * DD,
                                            nullptr, qh, MM, DD, DD, 0);
    gemm_f16_kernel<<<gD, 256, 0, stream>>>(xh, wk, bk + (size_t)l * DD,
                                            nullptr, kh, MM, DD, DD, 0);
    gemm_f16_kernel<<<gD, 256, 0, stream>>>(xh, wv, bv + (size_t)l * DD,
                                            nullptr, vh, MM, DD, DD, 0);
    v_transpose_kernel<<<vtBlocks, 256, 0, stream>>>(vh, vth);
    attn_kernel<<<gA, 32, 0, stream>>>(qh, kh, vth, attention_mask, ctxh);
    gemm_f16_kernel<<<gD, 256, 0, stream>>>(ctxh, wo, bo + (size_t)l * DD,
                                            tmp, nullptr, MM, DD, DD, 0);
    add_ln_kernel<<<MM, 256, 0, stream>>>(x, tmp, ln1_s + (size_t)l * DD,
                                          ln1_b + (size_t)l * DD, x, xh);
    gemm_f16_kernel<<<gF, 256, 0, stream>>>(xh, w1, b1 + (size_t)l * FF_,
                                            nullptr, hh, MM, FF_, DD, 1);
    gemm_f16_kernel<<<gD, 256, 0, stream>>>(hh, w2, b2 + (size_t)l * DD,
                                            tmp, nullptr, MM, DD, FF_, 0);
    add_ln_kernel<<<MM, 256, 0, stream>>>(x, tmp, ln2_s + (size_t)l * DD,
                                          ln2_b + (size_t)l * DD, x, xh);
  }

  // ---- classification head ----
  cls_pre_kernel<<<BB, 256, 0, stream>>>(x, pre_W, pre_b, pre_out);
  cls_out_kernel<<<1, 64, 0, stream>>>(pre_out, cls_W, cls_b, logits);
  (void)in_sizes; (void)n_in; (void)out_size; (void)ws_size;
}
